// LSWTTokenPoolerAttention_32220844654724
// MI455X (gfx1250) — matrix operand order repaired and checked by reading, not compiled, not run on hardware
//
#include <hip/hip_runtime.h>

#define D_MODEL 2048
#define N_HEADS 16
#define D_KEY   128
#define SEQ     2048
#define BATCH   2
#define M_ROWS  (BATCH * SEQ)          // 4096
#define KEY_SCALE 0.08838834764831845f // 128^-0.5

typedef __bf16 bf16_t;
typedef __attribute__((ext_vector_type(8)))  __bf16 v8bf;
typedef __attribute__((ext_vector_type(16))) __bf16 v16bf;
typedef __attribute__((ext_vector_type(8)))  float  v8f;

static __device__ __forceinline__ v16bf cat16(v8bf lo, v8bf hi) {
  return __builtin_shufflevector(lo, hi, 0,1,2,3,4,5,6,7,8,9,10,11,12,13,14,15);
}

// A-operand (16x32 bf16): lane m = lane&15, g = lane>>4.
// elems 0..7  = row[k0 + 8g + 0..7], elems 8..15 = row[k0 + 8g + 16..23]
static __device__ __forceinline__ v16bf load_a_frag(const bf16_t* rowp, int k0, int lane) {
  int ko = k0 + ((lane >> 4) << 3);
  v8bf lo = *(const v8bf*)(rowp + ko);
  v8bf hi = *(const v8bf*)(rowp + ko + 16);
  return cat16(lo, hi);
}

// B-operand (32x16 bf16): lane n = lane&15, g = lane>>4.
// elems 0..7  = col[k0 + 8g + 0..7], elems 8..15 = col[k0 + 8g + 16..23]  (columns contiguous: B^T row-major)
static __device__ __forceinline__ v16bf load_b_frag(const bf16_t* colp, int k0, int lane) {
  int ko = k0 + ((lane >> 4) << 3);
  v8bf lo = *(const v8bf*)(colp + ko);
  v8bf hi = *(const v8bf*)(colp + ko + 16);
  return cat16(lo, hi);
}

#define WMMA_BF16(a, b, c) \
  __builtin_amdgcn_wmma_f32_16x16x32_bf16(false, (a), false, (b), (short)0, (c), false, false)

// ---------------------------------------------------------------- converts
__global__ __launch_bounds__(256) void k_cvt_bf16(const float* __restrict__ in,
                                                  bf16_t* __restrict__ out, int n) {
  int i = blockIdx.x * 256 + threadIdx.x;
  if (i < n) out[i] = (bf16_t)in[i];
}

// in: fp32 [K=D_MODEL][N=D_MODEL]  ->  out: bf16 [N][K]
__global__ __launch_bounds__(256) void k_transpose_w(const float* __restrict__ in,
                                                     bf16_t* __restrict__ out) {
  __shared__ bf16_t tile[32][33];
  int c0 = blockIdx.x * 32, r0 = blockIdx.y * 32;
  int tx = threadIdx.x & 31, ty = threadIdx.x >> 5;  // 32 x 8
  for (int i = ty; i < 32; i += 8)
    tile[i][tx] = (bf16_t)in[(size_t)(r0 + i) * D_MODEL + c0 + tx];
  __syncthreads();
  for (int i = ty; i < 32; i += 8)
    out[(size_t)(c0 + i) * D_MODEL + r0 + tx] = tile[tx][i];
}

// ---------------------------------------------------------------- GEMM
#define ST_F32_PLAIN  0  // fp32 [M][N]
#define ST_BF16_HEADS 1  // bf16 [b,h,s,dk]
#define ST_BF16_VT    2  // bf16 [b,h,dk,s]

// Each wave computes a 32x64 output tile: 2 A-frags x 4 B-frags -> 8 WMMA/step.
// Ping/pong double-buffered over K: no register rotation copies.
__global__ __launch_bounds__(256) void k_gemm_bf16(
    const bf16_t* __restrict__ A, const bf16_t* __restrict__ Bt,
    const float* __restrict__ bias, float* __restrict__ Cf,
    bf16_t* __restrict__ Cb, int mode, float scale) {
  const int lane = threadIdx.x & 31;
  const int wid  = (blockIdx.x * 256 + threadIdx.x) >> 5;  // 8 waves/block
  const int NT = D_MODEL / 64;                             // 32
  const int mt = wid / NT, nt = wid % NT;
  const int m0 = mt * 32, n0 = nt * 64;
  const int l15 = lane & 15;

  const bf16_t* arow0 = A + (size_t)(m0 + l15) * D_MODEL;
  const bf16_t* arow1 = arow0 + (size_t)16 * D_MODEL;
  const bf16_t* brow  = Bt + (size_t)(n0 + l15) * D_MODEL;

  v8f acc[2][4] = {};
  v16bf a0p, a1p, bp[4], a0q, a1q, bq[4];

  auto loadset = [&](v16bf& x0, v16bf& x1, v16bf (&bb)[4], int k0) {
    x0 = load_a_frag(arow0, k0, lane);
    x1 = load_a_frag(arow1, k0, lane);
#pragma unroll
    for (int t = 0; t < 4; ++t)
      bb[t] = load_b_frag(brow + (size_t)t * 16 * D_MODEL, k0, lane);
  };
  auto mul = [&](const v16bf& x0, const v16bf& x1, const v16bf (&bb)[4]) {
#pragma unroll
    for (int t = 0; t < 4; ++t) {
      acc[0][t] = WMMA_BF16(x0, bb[t], acc[0][t]);
      acc[1][t] = WMMA_BF16(x1, bb[t], acc[1][t]);
    }
  };

  loadset(a0p, a1p, bp, 0);
  for (int k0 = 0; k0 < D_MODEL - 64; k0 += 64) {
    loadset(a0q, a1q, bq, k0 + 32);
    __builtin_prefetch(arow0 + k0 + 512, 0, 3);
    mul(a0p, a1p, bp);
    loadset(a0p, a1p, bp, k0 + 64);
    mul(a0q, a1q, bq);
  }
  loadset(a0q, a1q, bq, D_MODEL - 32);
  mul(a0p, a1p, bp);
  mul(a0q, a1q, bq);

  const int rofs = (lane >> 4) << 3;  // C layout: vgpr j -> row +rofs+j, col = tile+l15
#pragma unroll
  for (int t = 0; t < 4; ++t) {
    int col = n0 + t * 16 + l15;
    float bv = bias[col];
#pragma unroll
    for (int i = 0; i < 2; ++i) {
#pragma unroll
      for (int j = 0; j < 8; ++j) {
        int row = m0 + i * 16 + rofs + j;
        float v = (acc[i][t][j] + bv) * scale;
        if (mode == ST_F32_PLAIN) {
          Cf[(size_t)row * D_MODEL + col] = v;
        } else {
          int b = row >> 11, s = row & (SEQ - 1);
          int h = col >> 7,  dk = col & (D_KEY - 1);
          if (mode == ST_BF16_VT)
            Cb[(((size_t)((b * N_HEADS + h) * D_KEY + dk)) << 11) + s] = (bf16_t)v;
          else
            Cb[((((size_t)(b * N_HEADS + h) << 11) + s) * D_KEY) + dk] = (bf16_t)v;
        }
      }
    }
  }
}

// ---------------------------------------------------------------- gate
__global__ __launch_bounds__(256) void k_gate(const float* __restrict__ X,
                                              const float* __restrict__ wg,
                                              const float* __restrict__ bg,
                                              float* __restrict__ g) {
  int t = blockIdx.x * 256 + threadIdx.x;  // M_ROWS * N_HEADS
  int row = t >> 4, h = t & 15;
  const float* x = X + (size_t)row * D_MODEL;
  float acc = bg[h];
  for (int d = 0; d < D_MODEL; ++d) acc += x[d] * wg[d * N_HEADS + h];
  g[t] = 1.0f / (1.0f + __expf(-acc));
}

// ---------------------------------------------------------------- attention
// one wave per (b, h, 16-query tile); two-pass (max, then exp + P@V) softmax
__global__ __launch_bounds__(128) void k_attention(
    const bf16_t* __restrict__ Q, const bf16_t* __restrict__ K,
    const bf16_t* __restrict__ Vt, const float* __restrict__ bias_mask,
    const float* __restrict__ gate, bf16_t* __restrict__ O) {
  const int lane  = threadIdx.x & 31;
  const int wslot = threadIdx.x >> 5;
  const int wid = blockIdx.x * 4 + wslot;  // 0..4095
  const int qt = wid & 127;
  const int h  = (wid >> 7) & (N_HEADS - 1);
  const int b  = wid >> 11;
  const int q0 = qt * 16;
  const int l15  = lane & 15;
  const int rofs = (lane >> 4) << 3;

  const bf16_t* Qbh = Q  + (size_t)((b * N_HEADS + h) * SEQ) * D_KEY;
  const bf16_t* Kbh = K  + (size_t)((b * N_HEADS + h) * SEQ) * D_KEY;
  const bf16_t* Vbh = Vt + (size_t)((b * N_HEADS + h) * D_KEY) * SEQ;
  const float*  Bm  = bias_mask + (size_t)b * SEQ * SEQ;
  const float hscale = exp2f(-(float)(h + 1) * (8.0f / (float)N_HEADS));

  // preload Q A-fragments (16 rows x 128 dk); KEY_SCALE already folded into Q
  const bf16_t* qrow = Qbh + (size_t)(q0 + l15) * D_KEY;
  v16bf qa[4];
#pragma unroll
  for (int i = 0; i < 4; ++i) qa[i] = load_a_frag(qrow, i * 32, lane);

  auto loadk = [&](v16bf (&kk)[4], int kt) {
    const bf16_t* krow = Kbh + (size_t)(kt * 16 + l15) * D_KEY;
#pragma unroll
    for (int i = 0; i < 4; ++i) kk[i] = load_b_frag(krow, i * 32, lane);
  };
  auto score = [&](const v16bf (&kk)[4]) {
    v8f c = {};
#pragma unroll
    for (int i = 0; i < 4; ++i) c = WMMA_BF16(qa[i], kk[i], c);
    return c;
  };

  // ---- pass 1: per-row max of (scores + bias), ping/pong over key tiles ----
  v8f mx;
#pragma unroll
  for (int j = 0; j < 8; ++j) mx[j] = -3.0e38f;
  auto upd_max = [&](v8f c, int kt) {
#pragma unroll
    for (int j = 0; j < 8; ++j) {
      float s = c[j] + hscale * Bm[(size_t)(q0 + rofs + j) * SEQ + kt * 16 + l15];
      mx[j] = fmaxf(mx[j], s);
    }
  };
  {
    v16bf kp[4], kq[4];
    loadk(kp, 0);
    for (int kt = 0; kt < SEQ / 16 - 2; kt += 2) {
      loadk(kq, kt + 1);
      upd_max(score(kp), kt);
      loadk(kp, kt + 2);
      upd_max(score(kq), kt + 1);
    }
    loadk(kq, SEQ / 16 - 1);
    upd_max(score(kp), SEQ / 16 - 2);
    upd_max(score(kq), SEQ / 16 - 1);
  }
#pragma unroll
  for (int off = 1; off < 16; off <<= 1)
#pragma unroll
    for (int j = 0; j < 8; ++j)
      mx[j] = fmaxf(mx[j], __shfl_xor(mx[j], off, 32));

  // ---- pass 2: exp, sum, and O += P @ V (32 keys per step) ----
  __shared__ alignas(16) bf16_t pst[4][16][40];  // per-wave 16x32 P stage (+pad)
  v8f sum = {};
  v8f oacc[8] = {};
  for (int k2 = 0; k2 < SEQ / 32; ++k2) {
    // hoist all fragments for this step: K tiles for both halves + 8 V tiles
    v16bf kA[4], kB[4], vf[8];
    loadk(kA, k2 * 2);
    loadk(kB, k2 * 2 + 1);
#pragma unroll
    for (int d = 0; d < 8; ++d)
      vf[d] = load_b_frag(Vbh + (size_t)(d * 16 + l15) * SEQ, k2 * 32, lane);

#pragma unroll
    for (int half = 0; half < 2; ++half) {
      int kt = k2 * 2 + half;
      v8f c = half ? score(kB) : score(kA);
#pragma unroll
      for (int j = 0; j < 8; ++j) {
        float s = c[j] + hscale * Bm[(size_t)(q0 + rofs + j) * SEQ + kt * 16 + l15];
        float p = __expf(s - mx[j]);
        sum[j] += p;
        pst[wslot][rofs + j][half * 16 + l15] = (bf16_t)p;
      }
    }
    // same-wave LDS ops are in-order: stores above are visible to this read
    v16bf pa = load_a_frag(&pst[wslot][l15][0], 0, lane);
#pragma unroll
    for (int d = 0; d < 8; ++d) oacc[d] = WMMA_BF16(pa, vf[d], oacc[d]);
  }
#pragma unroll
  for (int off = 1; off < 16; off <<= 1)
#pragma unroll
    for (int j = 0; j < 8; ++j)
      sum[j] += __shfl_xor(sum[j], off, 32);

  // normalize, apply head gate, store bf16 O[b*S+q][h*128+dk]
#pragma unroll
  for (int j = 0; j < 8; ++j) {
    int q = q0 + rofs + j;
    float gv = gate[(size_t)((b << 11) + q) * N_HEADS + h];
    float inv = gv / sum[j];
#pragma unroll
    for (int d = 0; d < 8; ++d)
      O[(size_t)((b << 11) + q) * D_MODEL + h * D_KEY + d * 16 + l15] =
          (bf16_t)(oacc[d][j] * inv);
  }
}

// ---------------------------------------------------------------- launcher
extern "C" void kernel_launch(void* const* d_in, const int* in_sizes, int n_in,
                              void* d_out, int out_size, void* d_ws, size_t ws_size,
                              hipStream_t stream) {
  const float* states    = (const float*)d_in[0];
  const float* bias_mask = (const float*)d_in[1];
  const float* wq = (const float*)d_in[2];
  const float* bq = (const float*)d_in[3];
  const float* wk = (const float*)d_in[4];
  const float* bk = (const float*)d_in[5];
  const float* wv = (const float*)d_in[6];
  const float* bv = (const float*)d_in[7];
  const float* wg = (const float*)d_in[8];
  const float* bg = (const float*)d_in[9];
  const float* wo = (const float*)d_in[10];
  const float* bo = (const float*)d_in[11];
  float* out = (float*)d_out;

  char* ws = (char*)d_ws;
  auto take = [&](size_t bytes) {
    char* p = ws;
    ws += (bytes + 255) & ~(size_t)255;
    return p;
  };
  bf16_t* Xb   = (bf16_t*)take((size_t)M_ROWS * D_MODEL * 2);
  bf16_t* WqT  = (bf16_t*)take((size_t)D_MODEL * D_MODEL * 2);
  bf16_t* WkT  = (bf16_t*)take((size_t)D_MODEL * D_MODEL * 2);
  bf16_t* WvT  = (bf16_t*)take((size_t)D_MODEL * D_MODEL * 2);
  bf16_t* WoT  = (bf16_t*)take((size_t)D_MODEL * D_MODEL * 2);
  bf16_t* Qb   = (bf16_t*)take((size_t)M_ROWS * D_MODEL * 2);
  bf16_t* Kb   = (bf16_t*)take((size_t)M_ROWS * D_MODEL * 2);
  bf16_t* Vtb  = (bf16_t*)take((size_t)M_ROWS * D_MODEL * 2);
  bf16_t* Ob   = (bf16_t*)take((size_t)M_ROWS * D_MODEL * 2);
  float*  gbuf = (float*)take((size_t)M_ROWS * N_HEADS * 4);

  k_cvt_bf16<<<(M_ROWS * D_MODEL) / 256, 256, 0, stream>>>(states, Xb, M_ROWS * D_MODEL);

  dim3 tg(D_MODEL / 32, D_MODEL / 32);
  k_transpose_w<<<tg, 256, 0, stream>>>(wq, WqT);
  k_transpose_w<<<tg, 256, 0, stream>>>(wk, WkT);
  k_transpose_w<<<tg, 256, 0, stream>>>(wv, WvT);
  k_transpose_w<<<tg, 256, 0, stream>>>(wo, WoT);

  const int gemm_blocks = (M_ROWS / 32) * (D_MODEL / 64) / 8;  // 512
  k_gemm_bf16<<<gemm_blocks, 256, 0, stream>>>(Xb, WqT, bq, nullptr, Qb, ST_BF16_HEADS, KEY_SCALE);
  k_gemm_bf16<<<gemm_blocks, 256, 0, stream>>>(Xb, WkT, bk, nullptr, Kb, ST_BF16_HEADS, 1.0f);
  k_gemm_bf16<<<gemm_blocks, 256, 0, stream>>>(Xb, WvT, bv, nullptr, Vtb, ST_BF16_VT, 1.0f);

  k_gate<<<(M_ROWS * N_HEADS) / 256, 256, 0, stream>>>(states, wg, bg, gbuf);

  k_attention<<<(BATCH * N_HEADS * (SEQ / 16)) / 4, 128, 0, stream>>>(
      Qb, Kb, Vtb, bias_mask, gbuf, Ob);

  k_gemm_bf16<<<gemm_blocks, 256, 0, stream>>>(Ob, WoT, bo, out, nullptr, ST_F32_PLAIN, 1.0f);
}
